// LigandEncoderModule_67877663146006
// MI455X (gfx1250) — matrix operand order repaired and checked by reading, not compiled, not run on hardware
//
#include <hip/hip_runtime.h>

// ---------------------------------------------------------------------------
// MI455X (gfx1250) ligand-encoder. Edge GEMMs use V_WMMA_F32_16X16X32_BF16
// (fp32 accumulate) with 4-way M-blocking (64 edges / block) for B-fragment
// register reuse. Async global->LDS staging (ASYNCcnt path) when available.
// wave32 everywhere.
// ---------------------------------------------------------------------------

typedef __bf16 bf16;
typedef __attribute__((ext_vector_type(16))) bf16 bf16x16;
typedef __attribute__((ext_vector_type(8)))  bf16 bf16x8;
typedef __attribute__((ext_vector_type(8)))  float f32x8;

union ABFrag { bf16x16 v; bf16x8 h[2]; };

constexpr int NN = 40000;
constexpr int NE = 640000;
constexpr int D  = 128;
constexpr int MT = 4;          // M tiles (of 16 edges) per block
constexpr int MB = MT * 16;    // 64 edges per block

#if defined(__AMDGCN__) && __has_builtin(__builtin_amdgcn_global_load_async_to_lds_b128)
#define HAVE_ASYNC_LDS 1
#else
#define HAVE_ASYNC_LDS 0
#endif

#if HAVE_ASYNC_LDS
typedef int v4i_t __attribute__((__vector_size__(16)));
typedef __attribute__((address_space(1))) v4i_t* as1_v4i_p;   // global int4*
typedef __attribute__((address_space(3))) v4i_t* as3_v4i_p;   // LDS int4*
#endif

__device__ __forceinline__ void stage16(const uint4* g, uint4* lds_dst) {
#if HAVE_ASYNC_LDS
  __builtin_amdgcn_global_load_async_to_lds_b128((as1_v4i_p)g, (as3_v4i_p)lds_dst,
                                                 0, 0);
#else
  *lds_dst = *g;
#endif
}

__device__ __forceinline__ void stage_wait() {
#if HAVE_ASYNC_LDS
#if __has_builtin(__builtin_amdgcn_s_wait_asynccnt)
  __builtin_amdgcn_s_wait_asynccnt(0);
#else
  asm volatile("s_wait_asynccnt 0" ::: "memory");
#endif
#endif
}

__device__ __forceinline__ float wave_sum32(float x) {
  #pragma unroll
  for (int m = 16; m >= 1; m >>= 1) x += __shfl_xor(x, m, 32);
  return x;
}

// ---------------------------------------------------------------------------
__global__ void k_fill_f32(float* __restrict__ p, float v, int n) {
  int i = blockIdx.x * blockDim.x + threadIdx.x;
  if (i < n) p[i] = v;
}

// s0 = nodes @ w_in + b_in   (40000x23 @ 23x128)
__global__ void k_node_embed(const float* __restrict__ nodes,
                             const float* __restrict__ w_in,
                             const float* __restrict__ b_in,
                             float* __restrict__ s) {
  int i = blockIdx.x * blockDim.x + threadIdx.x;
  if (i >= NN * D) return;
  int n = i >> 7, c = i & 127;
  const float* row = nodes + n * 23;
  float acc = b_in[c];
  #pragma unroll
  for (int k = 0; k < 23; k++) acc += row[k] * w_in[k * D + c];
  s[i] = acc;
}

// segment_sum of coords[src] by dst + counts
__global__ void k_coord_scatter(const int* __restrict__ ei,
                                const float* __restrict__ coords,
                                float* __restrict__ nbr, float* __restrict__ cnt) {
  int e = blockIdx.x * blockDim.x + threadIdx.x;
  if (e >= NE) return;
  int src = ei[e], dst = ei[NE + e];
  atomicAdd(&cnt[dst], 1.0f);
  #pragma unroll
  for (int c = 0; c < 3; c++) atomicAdd(&nbr[dst * 3 + c], coords[src * 3 + c]);
}

// GVP node block: one node per 128-thread block
__global__ void __launch_bounds__(128) k_gvp(
    const float* __restrict__ coords, const float* __restrict__ nbr,
    const float* __restrict__ cnt, const float* __restrict__ s_in,
    const float* __restrict__ wh,  const float* __restrict__ wsmat,
    const float* __restrict__ bs,  const float* __restrict__ wv,
    const float* __restrict__ wsv, const float* __restrict__ bsv,
    float* __restrict__ s_out, bf16* __restrict__ s_bf, float* __restrict__ v_out) {
  int n = blockIdx.x, t = threadIdx.x;
  __shared__ float xin[132];   // [s row | vn(4)]
  __shared__ float s2s[D];
  __shared__ float ivs[3];
  __shared__ float gates[4];
  if (t < 3) {
    float c_ = fmaxf(cnt[n], 1.0f);
    ivs[t] = nbr[n * 3 + t] / c_ - coords[n * 3 + t];
  }
  __syncthreads();
  if (t == 0) {
    float nn2 = ivs[0]*ivs[0] + ivs[1]*ivs[1] + ivs[2]*ivs[2];
    float inv = 1.0f / sqrtf(fmaxf(nn2, 1e-8f));
    ivs[0] *= inv; ivs[1] *= inv; ivs[2] *= inv;
  }
  xin[t] = s_in[n * D + t];
  __syncthreads();
  if (t < 4) {
    float w = wh[t];
    float sq = w * w * (ivs[0]*ivs[0] + ivs[1]*ivs[1] + ivs[2]*ivs[2]);
    xin[D + t] = sqrtf(fmaxf(sq, 1e-8f));
  }
  __syncthreads();
  float acc = bs[t];
  for (int k = 0; k < 132; k++) acc += xin[k] * wsmat[k * D + t];
  float r = fmaxf(acc, 0.0f);
  s2s[t] = r;
  __syncthreads();
  if (t < 4) {
    float g = bsv[t];
    for (int k = 0; k < D; k++) g += s2s[k] * wsv[k * 4 + t];
    gates[t] = 1.0f / (1.0f + __expf(-g));
  }
  s_out[n * D + t] = r;
  s_bf[n * D + t]  = (bf16)r;
  __syncthreads();
  if (t < 12) {
    int o = t / 3, c = t % 3;
    float vo = 0.0f;
    #pragma unroll
    for (int h = 0; h < 4; h++) vo += wh[h] * ivs[c] * wv[h * 4 + o];
    v_out[n * 12 + t] = vo * gates[o];
  }
}

// e = rbf(dist) @ w_e + b_e ; one edge per 128-thread block
__global__ void __launch_bounds__(128) k_rbf(
    const float* __restrict__ dist, const float* __restrict__ w_e,
    const float* __restrict__ b_e, float* __restrict__ e, bf16* __restrict__ e_bf) {
  int eid = blockIdx.x, t = threadIdx.x;
  __shared__ float rbf[16];
  if (t < 16) {
    float mu  = 1.0f * (float)t;              // linspace(0,15,16), step 1
    float d   = (dist[eid] - mu) * (16.0f / 15.0f);
    rbf[t] = __expf(-d * d) + 1e-4f;
  }
  __syncthreads();
  float acc = b_e[t];
  #pragma unroll
  for (int k = 0; k < 16; k++) acc += rbf[k] * w_e[k * D + t];
  e[(size_t)eid * D + t]    = acc;
  e_bf[(size_t)eid * D + t] = (bf16)acc;
}

// transpose (layers, K, 128) fp32 -> (layers, 128, K) bf16
__global__ void k_w2bf_T(const float* __restrict__ src, bf16* __restrict__ dst,
                         int Ksz, int layers) {
  int i = blockIdx.x * blockDim.x + threadIdx.x;
  int tot = layers * Ksz * D;
  if (i >= tot) return;
  int l = i / (Ksz * D), rem = i % (Ksz * D);
  int n = rem / Ksz, k = rem % Ksz;
  dst[i] = (bf16)src[((size_t)l * Ksz + k) * D + n];
}

// ---------------------------------------------------------------------------
// WMMA attention-logit kernel: 64-edge tile, A = [s_dst|s_src|e] (64x384 bf16)
// 8 waves x (4 M-tiles x 16 N) ; B fragment reused across the 4 M-tiles.
// ---------------------------------------------------------------------------
__global__ void __launch_bounds__(256) k_edge_logit(
    const int* __restrict__ ei, const bf16* __restrict__ s_bf,
    const bf16* __restrict__ e_bf, const bf16* __restrict__ WaT,
    const float* __restrict__ ba, const float* __restrict__ avec,
    float* __restrict__ logit) {
  constexpr int K = 384;
  __shared__ __align__(16) bf16 A[MB * K];
  __shared__ int dsts[MB], srcs[MB];
  __shared__ float lg[MB];
  int tid = threadIdx.x;
  int e0  = blockIdx.x * MB;
  if (tid < MB) {
    srcs[tid] = ei[e0 + tid];
    dsts[tid] = ei[NE + e0 + tid];
    lg[tid]   = 0.0f;
    if (e0 + MB < NE)                   // prefetch next tile's streaming rows
      __builtin_prefetch(e_bf + (size_t)(e0 + MB + tid) * D, 0, 1);
  }
  __syncthreads();
  uint4* A4 = (uint4*)A;
  for (int i = tid; i < MB * 48; i += 256) {
    int r = i / 48, p = i % 48;
    const uint4* q;
    if (p < 16)      q = (const uint4*)(s_bf + (size_t)dsts[r] * D) + p;
    else if (p < 32) q = (const uint4*)(s_bf + (size_t)srcs[r] * D) + (p - 16);
    else             q = (const uint4*)(e_bf + (size_t)(e0 + r) * D) + (p - 32);
    stage16(q, &A4[r * 48 + p]);
  }
  stage_wait();
  __syncthreads();
  int lane = tid & 31, wv = tid >> 5;
  int m = lane & 15, half = lane >> 4;
  int n = wv * 16 + m;                  // C-frag column (ISA: N = lane%16)
  f32x8 c[MT] = {};
  const bf16* Brow = WaT + (size_t)n * K;
  for (int kk = 0; kk < K; kk += 32) {
    ABFrag b;
    b.h[0] = *(const bf16x8*)(Brow + kk + 8 * half);
    b.h[1] = *(const bf16x8*)(Brow + kk + 16 + 8 * half);
    #pragma unroll
    for (int t = 0; t < MT; t++) {
      ABFrag a;
      const bf16* Arow = A + (t * 16 + m) * K;
      a.h[0] = *(const bf16x8*)(Arow + kk + 8 * half);
      a.h[1] = *(const bf16x8*)(Arow + kk + 16 + 8 * half);
      c[t] = __builtin_amdgcn_wmma_f32_16x16x32_bf16(false, a.v, false, b.v,
                                                     (short)0, c[t], false, false);
    }
  }
  float bn = ba[n], an = avec[n];
  #pragma unroll
  for (int t = 0; t < MT; t++) {
    #pragma unroll
    for (int r = 0; r < 8; r++) {
      int mr = t * 16 + half * 8 + r;   // C-frag row (ISA: M = r + 8*hi)
      float h = c[t][r] + bn;
      h = (h > 0.0f) ? h : 0.2f * h;    // leaky_relu 0.2
      atomicAdd(&lg[mr], h * an);
    }
  }
  __syncthreads();
  if (tid < MB) logit[e0 + tid] = lg[tid];
}

// segment max via float-as-int atomic trick (mx pre-filled with -inf)
__global__ void k_seg_max(const int* __restrict__ ei, const float* __restrict__ logit,
                          float* __restrict__ mx) {
  int e = blockIdx.x * blockDim.x + threadIdx.x;
  if (e >= NE) return;
  int dst = ei[NE + e];
  float v = logit[e];
  if (v >= 0.0f) atomicMax((int*)&mx[dst], __float_as_int(v));
  else           atomicMin((unsigned int*)&mx[dst], __float_as_uint(v));
}

__global__ void k_seg_sumexp(const int* __restrict__ ei, const float* __restrict__ logit,
                             const float* __restrict__ mx, float* __restrict__ z) {
  int e = blockIdx.x * blockDim.x + threadIdx.x;
  if (e >= NE) return;
  int dst = ei[NE + e];
  atomicAdd(&z[dst], __expf(logit[e] - mx[dst]));
}

// ---------------------------------------------------------------------------
// WMMA message kernel: msg = [s_src|e] @ Wm + bm, scatter alpha*msg into s_agg
// ---------------------------------------------------------------------------
__global__ void __launch_bounds__(256) k_edge_msg_agg(
    const int* __restrict__ ei, const bf16* __restrict__ s_bf,
    const bf16* __restrict__ e_bf, const bf16* __restrict__ WmT,
    const float* __restrict__ bm, const float* __restrict__ logit,
    const float* __restrict__ mx, const float* __restrict__ z,
    float* __restrict__ s_agg) {
  constexpr int K = 256;
  __shared__ __align__(16) bf16 A[MB * K];
  __shared__ int dsts[MB];
  __shared__ int srcs[MB];
  __shared__ float alph[MB];
  int tid = threadIdx.x;
  int e0  = blockIdx.x * MB;
  if (tid < MB) {
    int s_ = ei[e0 + tid], d_ = ei[NE + e0 + tid];
    srcs[tid] = s_; dsts[tid] = d_;
    alph[tid] = __expf(logit[e0 + tid] - mx[d_]) / (z[d_] + 1e-9f);
    if (e0 + MB < NE)
      __builtin_prefetch(e_bf + (size_t)(e0 + MB + tid) * D, 0, 1);
  }
  __syncthreads();
  uint4* A4 = (uint4*)A;
  for (int i = tid; i < MB * 32; i += 256) {
    int r = i / 32, p = i % 32;
    const uint4* q = (p < 16)
        ? (const uint4*)(s_bf + (size_t)srcs[r] * D) + p
        : (const uint4*)(e_bf + (size_t)(e0 + r) * D) + (p - 16);
    stage16(q, &A4[r * 32 + p]);
  }
  stage_wait();
  __syncthreads();
  int lane = tid & 31, wv = tid >> 5;
  int m = lane & 15, half = lane >> 4;
  int n = wv * 16 + m;
  f32x8 c[MT] = {};
  const bf16* Brow = WmT + (size_t)n * K;
  for (int kk = 0; kk < K; kk += 32) {
    ABFrag b;
    b.h[0] = *(const bf16x8*)(Brow + kk + 8 * half);
    b.h[1] = *(const bf16x8*)(Brow + kk + 16 + 8 * half);
    #pragma unroll
    for (int t = 0; t < MT; t++) {
      ABFrag a;
      const bf16* Arow = A + (t * 16 + m) * K;
      a.h[0] = *(const bf16x8*)(Arow + kk + 8 * half);
      a.h[1] = *(const bf16x8*)(Arow + kk + 16 + 8 * half);
      c[t] = __builtin_amdgcn_wmma_f32_16x16x32_bf16(false, a.v, false, b.v,
                                                     (short)0, c[t], false, false);
    }
  }
  float bn = bm[n];
  #pragma unroll
  for (int t = 0; t < MT; t++) {
    #pragma unroll
    for (int r = 0; r < 8; r++) {
      int mr = t * 16 + half * 8 + r;
      float msg = c[t][r] + bn;
      atomicAdd(&s_agg[(size_t)dsts[mr] * D + n], alph[mr] * msg);
    }
  }
}

// vector messages: vagg[dst] += alpha * (v[src] x Wvm)
__global__ void k_vmsg(const int* __restrict__ ei, const float* __restrict__ v,
                       const float* __restrict__ Wvm, const float* __restrict__ logit,
                       const float* __restrict__ mx, const float* __restrict__ z,
                       float* __restrict__ vagg) {
  int i = blockIdx.x * blockDim.x + threadIdx.x;
  if (i >= NE * 12) return;
  int e = i / 12, j = i % 12, w = j / 3, cc = j % 3;
  int s_ = ei[e], d_ = ei[NE + e];
  float al = __expf(logit[e] - mx[d_]) / (z[d_] + 1e-9f);
  float acc = 0.0f;
  #pragma unroll
  for (int h = 0; h < 4; h++) acc += v[(size_t)s_ * 12 + h * 3 + cc] * Wvm[h * 4 + w];
  atomicAdd(&vagg[(size_t)d_ * 12 + j], al * acc);
}

// node update: s = LN(s+agg); vgate = sigmoid(s@Wg+bg); v += vgate*vagg
__global__ void __launch_bounds__(256) k_node_update(
    const float* __restrict__ s_agg, const float* __restrict__ vagg,
    const float* __restrict__ ln_g, const float* __restrict__ ln_b,
    const float* __restrict__ Wg, const float* __restrict__ bg,
    float* __restrict__ s, bf16* __restrict__ s_bf, float* __restrict__ v) {
  int node = blockIdx.x * 8 + (threadIdx.x >> 5);
  if (node >= NN) return;
  int lane = threadIdx.x & 31;
  float x[4], sum = 0.0f;
  #pragma unroll
  for (int j = 0; j < 4; j++) {
    int c = lane + 32 * j;
    x[j] = s[(size_t)node * D + c] + s_agg[(size_t)node * D + c];
    sum += x[j];
  }
  float mean = wave_sum32(sum) * (1.0f / 128.0f);
  float vs = 0.0f;
  #pragma unroll
  for (int j = 0; j < 4; j++) { float d = x[j] - mean; vs += d * d; }
  float inv = rsqrtf(wave_sum32(vs) * (1.0f / 128.0f) + 1e-5f);
  float po[4] = {0.f, 0.f, 0.f, 0.f};
  #pragma unroll
  for (int j = 0; j < 4; j++) {
    int c = lane + 32 * j;
    float sn = ln_g[c] * (x[j] - mean) * inv + ln_b[c];
    s[(size_t)node * D + c]    = sn;
    s_bf[(size_t)node * D + c] = (bf16)sn;
    #pragma unroll
    for (int o = 0; o < 4; o++) po[o] += sn * Wg[c * 4 + o];
  }
  #pragma unroll
  for (int o = 0; o < 4; o++) po[o] = wave_sum32(po[o]);
  if (lane < 12) {
    int o = lane / 3;
    float g = 1.0f / (1.0f + __expf(-(po[o] + bg[o])));
    v[(size_t)node * 12 + lane] += g * vagg[(size_t)node * 12 + lane];
  }
}

// ---------------------------------------------------------------------------
// WMMA edge update: e = LN(e + [s_src|s_dst|e] @ We + be)
// ---------------------------------------------------------------------------
__global__ void __launch_bounds__(256) k_edge_update(
    const int* __restrict__ ei, const bf16* __restrict__ s_bf,
    bf16* __restrict__ e_bf, float* __restrict__ e,
    const bf16* __restrict__ WeT, const float* __restrict__ be,
    const float* __restrict__ g2, const float* __restrict__ b2) {
  constexpr int K = 384;
  __shared__ __align__(16) bf16 A[MB * K];
  __shared__ int dsts[MB], srcs[MB];
  __shared__ float rs[MB], rq[MB];
  int tid = threadIdx.x;
  int e0  = blockIdx.x * MB;
  if (tid < MB) {
    srcs[tid] = ei[e0 + tid];
    dsts[tid] = ei[NE + e0 + tid];
    rs[tid] = 0.0f; rq[tid] = 0.0f;
    if (e0 + MB < NE)
      __builtin_prefetch(e_bf + (size_t)(e0 + MB + tid) * D, 0, 1);
  }
  __syncthreads();
  uint4* A4 = (uint4*)A;
  for (int i = tid; i < MB * 48; i += 256) {
    int r = i / 48, p = i % 48;
    const uint4* q;
    if (p < 16)      q = (const uint4*)(s_bf + (size_t)srcs[r] * D) + p;   // src first
    else if (p < 32) q = (const uint4*)(s_bf + (size_t)dsts[r] * D) + (p - 16);
    else             q = (const uint4*)(e_bf + (size_t)(e0 + r) * D) + (p - 32);
    stage16(q, &A4[r * 48 + p]);
  }
  stage_wait();
  __syncthreads();
  int lane = tid & 31, wv = tid >> 5;
  int m = lane & 15, half = lane >> 4;
  int n = wv * 16 + m;
  f32x8 c[MT] = {};
  const bf16* Brow = WeT + (size_t)n * K;
  for (int kk = 0; kk < K; kk += 32) {
    ABFrag b;
    b.h[0] = *(const bf16x8*)(Brow + kk + 8 * half);
    b.h[1] = *(const bf16x8*)(Brow + kk + 16 + 8 * half);
    #pragma unroll
    for (int t = 0; t < MT; t++) {
      ABFrag a;
      const bf16* Arow = A + (t * 16 + m) * K;
      a.h[0] = *(const bf16x8*)(Arow + kk + 8 * half);
      a.h[1] = *(const bf16x8*)(Arow + kk + 16 + 8 * half);
      c[t] = __builtin_amdgcn_wmma_f32_16x16x32_bf16(false, a.v, false, b.v,
                                                     (short)0, c[t], false, false);
    }
  }
  float bn = be[n];
  float tv[MT][8];
  #pragma unroll
  for (int t = 0; t < MT; t++) {
    #pragma unroll
    for (int r = 0; r < 8; r++) {
      int mr = t * 16 + half * 8 + r;
      float u = c[t][r] + bn + e[(size_t)(e0 + mr) * D + n];
      tv[t][r] = u;
      atomicAdd(&rs[mr], u);
      atomicAdd(&rq[mr], u * u);
    }
  }
  __syncthreads();
  float gn = g2[n], bnn = b2[n];
  #pragma unroll
  for (int t = 0; t < MT; t++) {
    #pragma unroll
    for (int r = 0; r < 8; r++) {
      int mr = t * 16 + half * 8 + r;
      float mean = rs[mr] * (1.0f / 128.0f);
      float var  = rq[mr] * (1.0f / 128.0f) - mean * mean;
      float out  = gn * (tv[t][r] - mean) * rsqrtf(var + 1e-5f) + bnn;
      e[(size_t)(e0 + mr) * D + n]    = out;
      e_bf[(size_t)(e0 + mr) * D + n] = (bf16)out;
    }
  }
}

__global__ void k_copy_out(const float* __restrict__ s, const float* __restrict__ v,
                           float* __restrict__ out) {
  int i = blockIdx.x * blockDim.x + threadIdx.x;
  constexpr int SN = NN * D;
  constexpr int TOT = SN + NN * 12;
  if (i < SN) out[i] = s[i];
  else if (i < TOT) out[i] = v[i - SN];
}

// ---------------------------------------------------------------------------
extern "C" void kernel_launch(void* const* d_in, const int* in_sizes, int n_in,
                              void* d_out, int out_size, void* d_ws, size_t ws_size,
                              hipStream_t stream) {
  (void)in_sizes; (void)n_in; (void)out_size; (void)ws_size;
  const float* nodes  = (const float*)d_in[0];
  const float* coords = (const float*)d_in[1];
  const int*   ei     = (const int*)d_in[2];
  const float* dist   = (const float*)d_in[3];
  const float* w_in   = (const float*)d_in[4];
  const float* b_in   = (const float*)d_in[5];
  const float* gvp_wh = (const float*)d_in[6];
  const float* gvp_ws = (const float*)d_in[7];
  const float* gvp_bs = (const float*)d_in[8];
  const float* gvp_wv = (const float*)d_in[9];
  const float* gvp_wsv= (const float*)d_in[10];
  const float* gvp_bsv= (const float*)d_in[11];
  const float* w_e    = (const float*)d_in[12];
  const float* b_e    = (const float*)d_in[13];
  const float* Wa     = (const float*)d_in[14];
  const float* ba     = (const float*)d_in[15];
  const float* avec   = (const float*)d_in[16];
  const float* Wm     = (const float*)d_in[17];
  const float* bm     = (const float*)d_in[18];
  const float* Wvm    = (const float*)d_in[19];
  const float* Wg     = (const float*)d_in[20];
  const float* bg     = (const float*)d_in[21];
  const float* ln1_g  = (const float*)d_in[22];
  const float* ln1_b  = (const float*)d_in[23];
  const float* We     = (const float*)d_in[24];
  const float* be     = (const float*)d_in[25];
  const float* ln2_g  = (const float*)d_in[26];
  const float* ln2_b  = (const float*)d_in[27];

  char* ws = (char*)d_ws;
  size_t off = 0;
  auto alloc = [&](size_t bytes) -> char* {
    char* p = ws + off;
    off = (off + bytes + 255) & ~(size_t)255;
    return p;
  };
  float* s      = (float*)alloc((size_t)NN * D * 4);
  bf16*  s_bf   = (bf16*) alloc((size_t)NN * D * 2);
  float* v      = (float*)alloc((size_t)NN * 12 * 4);
  float* nbr    = (float*)alloc((size_t)NN * 3 * 4);
  float* cnt    = (float*)alloc((size_t)NN * 4);
  float* e      = (float*)alloc((size_t)NE * D * 4);
  bf16*  e_bf   = (bf16*) alloc((size_t)NE * D * 2);
  float* logit  = (float*)alloc((size_t)NE * 4);
  float* mx     = (float*)alloc((size_t)NN * 4);
  float* z      = (float*)alloc((size_t)NN * 4);
  float* s_agg  = (float*)alloc((size_t)NN * D * 4);
  float* vagg   = (float*)alloc((size_t)NN * 12 * 4);
  bf16*  WaT    = (bf16*) alloc((size_t)3 * D * 384 * 2);
  bf16*  WmT    = (bf16*) alloc((size_t)3 * D * 256 * 2);
  bf16*  WeT    = (bf16*) alloc((size_t)2 * D * 384 * 2);

  const int BS = 256;
  auto blocks = [](int n) { return dim3((unsigned)((n + 255) / 256)); };

  // prologue
  k_fill_f32<<<blocks(NN * 3), BS, 0, stream>>>(nbr, 0.0f, NN * 3);
  k_fill_f32<<<blocks(NN), BS, 0, stream>>>(cnt, 0.0f, NN);
  k_node_embed<<<blocks(NN * D), BS, 0, stream>>>(nodes, w_in, b_in, s);
  k_coord_scatter<<<blocks(NE), BS, 0, stream>>>(ei, coords, nbr, cnt);
  k_gvp<<<dim3(NN), 128, 0, stream>>>(coords, nbr, cnt, s, gvp_wh, gvp_ws, gvp_bs,
                                      gvp_wv, gvp_wsv, gvp_bsv, s, s_bf, v);
  k_rbf<<<dim3(NE), 128, 0, stream>>>(dist, w_e, b_e, e, e_bf);
  k_w2bf_T<<<blocks(3 * 384 * D), BS, 0, stream>>>(Wa, WaT, 384, 3);
  k_w2bf_T<<<blocks(3 * 256 * D), BS, 0, stream>>>(Wm, WmT, 256, 3);
  k_w2bf_T<<<blocks(2 * 384 * D), BS, 0, stream>>>(We, WeT, 384, 2);

  for (int l = 0; l < 3; l++) {
    k_fill_f32<<<blocks(NN), BS, 0, stream>>>(mx, -INFINITY, NN);
    k_fill_f32<<<blocks(NN), BS, 0, stream>>>(z, 0.0f, NN);
    k_fill_f32<<<blocks(NN * D), BS, 0, stream>>>(s_agg, 0.0f, NN * D);
    k_fill_f32<<<blocks(NN * 12), BS, 0, stream>>>(vagg, 0.0f, NN * 12);

    k_edge_logit<<<dim3(NE / MB), 256, 0, stream>>>(
        ei, s_bf, e_bf, WaT + (size_t)l * D * 384, ba + l * D, avec + l * D, logit);
    k_seg_max<<<blocks(NE), BS, 0, stream>>>(ei, logit, mx);
    k_seg_sumexp<<<blocks(NE), BS, 0, stream>>>(ei, logit, mx, z);
    k_edge_msg_agg<<<dim3(NE / MB), 256, 0, stream>>>(
        ei, s_bf, e_bf, WmT + (size_t)l * D * 256, bm + l * D, logit, mx, z, s_agg);
    k_vmsg<<<blocks(NE * 12), BS, 0, stream>>>(ei, v, Wvm + l * 16, logit, mx, z, vagg);
    k_node_update<<<dim3(NN / 8), 256, 0, stream>>>(
        s_agg, vagg, ln1_g + l * D, ln1_b + l * D, Wg + l * D * 4, bg + l * 4,
        s, s_bf, v);
    if (l < 2) {
      k_edge_update<<<dim3(NE / MB), 256, 0, stream>>>(
          ei, s_bf, e_bf, e, WeT + (size_t)l * D * 384, be + l * D,
          ln2_g + l * D, ln2_b + l * D);
    }
  }
  k_copy_out<<<blocks(NN * D + NN * 12), BS, 0, stream>>>(s, v, (float*)d_out);
}
